// LinearAttention_55697135894856
// MI455X (gfx1250) — matrix operand order repaired and checked by reading, not compiled, not run on hardware
//
#include <hip/hip_runtime.h>
#include <hip/hip_bf16.h>

typedef __attribute__((ext_vector_type(16))) __bf16 v16bf;
typedef __attribute__((ext_vector_type(8)))  float  v8f;

union BfPack {
  struct { uint4 lo, hi; } u;   // 32 bytes = 16 bf16 halves
  v16bf v;
};

#define BB     16
#define CC     64
#define NPIX   16384      // 128*128 spatial positions per batch
#define HIDD   128
#define NT     128        // positions per tile
#define NTILES 128        // NPIX / NT
#define M1     384        // 3*HID rows of qkv
#define QP     392        // padded pitch (halves) of transposed qkv tile [n][row]

// hardware packed f32->bf16 (RNE): one VALU op for two values
static __device__ __forceinline__ unsigned cvt_pk_bf16(float a, float b) {
  unsigned r;
  asm("v_cvt_pk_bf16_f32 %0, %1, %2" : "=v"(r) : "v"(a), "v"(b));
  return r;
}
static __device__ __forceinline__ unsigned short f2bf1(float a) {
  unsigned r;
  asm("v_cvt_pk_bf16_f32 %0, %1, 0" : "=v"(r) : "v"(a));
  return (unsigned short)r;
}
static __device__ __forceinline__ float bf2f(unsigned short h) {
  return __uint_as_float(((unsigned)h) << 16);
}

// ---------------------------------------------------------------------------
// K1: qkv tile GEMM (WMMA), softmax(q)*scale -> global q' (bf16 [n][hid]),
//     softmax(k); k'/v staged into WMMA A/B layouts; per-tile 32x32 context
//     computed with WMMA (K = 128 positions) -> ctxPartial.
// ---------------------------------------------------------------------------
__global__ __launch_bounds__(256) void k_qkv(const float* __restrict__ x,
                                             const float* __restrict__ wqkv,
                                             unsigned short* __restrict__ Qp,
                                             float* __restrict__ ctxPartial) {
  extern __shared__ char smem[];
  unsigned short* sW  = (unsigned short*)smem;              // 384*64      48KB
  unsigned short* sX  = (unsigned short*)(smem + 49152);    // [n][c]      16KB
  unsigned short* sT  = (unsigned short*)(smem + 65536);    // [n][392]    98KB (qkv^T)
  unsigned short* sKA = (unsigned short*)(smem + 165888);   // k' A-layout 32KB
  unsigned short* sVB = (unsigned short*)(smem + 198656);   // v  B-layout 32KB

  const int tid  = threadIdx.x;
  const int b    = blockIdx.y;
  const int tile = blockIdx.x;
  const int n0   = tile * NT;

  for (int i = tid; i < (M1 * CC) / 2; i += 256) {     // pairwise convert W
    float2 w2 = *(const float2*)(wqkv + 2 * i);
    ((unsigned*)sW)[i] = cvt_pk_bf16(w2.x, w2.y);
  }
  for (int i = tid; i < NT * CC; i += 256) {
    int c = i >> 7, j = i & (NT - 1);
    sX[j * CC + c] = f2bf1(x[(size_t)(b * CC + c) * NPIX + n0 + j]);
  }
  __syncthreads();

  const int wave = tid >> 5, lane = tid & 31;
  const int lg = lane >> 4, lr = lane & 15;
  const int nb = wave * 16;                 // wave owns one 16-column block

  for (int mt = 0; mt < 24; ++mt) {
    v8f acc = {0.f, 0.f, 0.f, 0.f, 0.f, 0.f, 0.f, 0.f};
    const unsigned short* ar = sW + (mt * 16 + lr) * CC;
    const unsigned short* br = sX + (nb + lr) * CC;
    for (int ks = 0; ks < 64; ks += 32) {
      BfPack a, bm;
      a.u.lo  = *(const uint4*)(ar + ks + 8 * lg);
      a.u.hi  = *(const uint4*)(ar + ks + 16 + 8 * lg);
      bm.u.lo = *(const uint4*)(br + ks + 8 * lg);
      bm.u.hi = *(const uint4*)(br + ks + 16 + 8 * lg);
      acc = __builtin_amdgcn_wmma_f32_16x16x32_bf16(false, a.v, false, bm.v,
                                                    (short)0, acc, false, false);
    }
    // D rows are contiguous per lane in transposed layout -> one b128 store
    uint4 pk;
    pk.x = cvt_pk_bf16(acc[0], acc[1]); pk.y = cvt_pk_bf16(acc[2], acc[3]);
    pk.z = cvt_pk_bf16(acc[4], acc[5]); pk.w = cvt_pk_bf16(acc[6], acc[7]);
    *(uint4*)(sT + (nb + lr) * QP + mt * 16 + 8 * lg) = pk;
  }
  __syncthreads();

  const float scale = 0.17677669529663687f;  // 32^-0.5
  for (int task = tid; task < NT * 4; task += 256) {
    int j = task & (NT - 1);
    int h = task >> 7;
    const int kstep = j >> 5, jj = j & 31;
    const int lgA = (jj >> 3) & 1, iA = (jj & 7) + 8 * (jj >> 4);
    const int lgB = (jj >> 3) & 1, iB = (jj & 7) + 8 * (jj >> 4);

    union { uint4 q[4]; unsigned short hh[32]; } u;
    {  // q softmax over d, scaled, to global (vector load/store)
      const unsigned short* qp = sT + j * QP + h * 32;
      u.q[0] = *(const uint4*)qp;        u.q[1] = *(const uint4*)(qp + 8);
      u.q[2] = *(const uint4*)(qp + 16); u.q[3] = *(const uint4*)(qp + 24);
      float mx = -1e30f;
      for (int d = 0; d < 32; ++d) mx = fmaxf(mx, bf2f(u.hh[d]));
      float ev[32], s = 0.f;
      for (int d = 0; d < 32; ++d) { ev[d] = __expf(bf2f(u.hh[d]) - mx); s += ev[d]; }
      float inv = scale / s;
      uint4 o0, o1, o2, o3;
      o0.x = cvt_pk_bf16(ev[0]*inv, ev[1]*inv);   o0.y = cvt_pk_bf16(ev[2]*inv, ev[3]*inv);
      o0.z = cvt_pk_bf16(ev[4]*inv, ev[5]*inv);   o0.w = cvt_pk_bf16(ev[6]*inv, ev[7]*inv);
      o1.x = cvt_pk_bf16(ev[8]*inv, ev[9]*inv);   o1.y = cvt_pk_bf16(ev[10]*inv, ev[11]*inv);
      o1.z = cvt_pk_bf16(ev[12]*inv, ev[13]*inv); o1.w = cvt_pk_bf16(ev[14]*inv, ev[15]*inv);
      o2.x = cvt_pk_bf16(ev[16]*inv, ev[17]*inv); o2.y = cvt_pk_bf16(ev[18]*inv, ev[19]*inv);
      o2.z = cvt_pk_bf16(ev[20]*inv, ev[21]*inv); o2.w = cvt_pk_bf16(ev[22]*inv, ev[23]*inv);
      o3.x = cvt_pk_bf16(ev[24]*inv, ev[25]*inv); o3.y = cvt_pk_bf16(ev[26]*inv, ev[27]*inv);
      o3.z = cvt_pk_bf16(ev[28]*inv, ev[29]*inv); o3.w = cvt_pk_bf16(ev[30]*inv, ev[31]*inv);
      unsigned short* qo = Qp + (size_t)(b * NPIX + n0 + j) * HIDD + h * 32;
      *(uint4*)qo = o0; *(uint4*)(qo + 8) = o1;
      *(uint4*)(qo + 16) = o2; *(uint4*)(qo + 24) = o3;
    }
    {  // k softmax over d; scatter k' into WMMA A-layout staging
      const unsigned short* kp = sT + j * QP + 128 + h * 32;
      u.q[0] = *(const uint4*)kp;        u.q[1] = *(const uint4*)(kp + 8);
      u.q[2] = *(const uint4*)(kp + 16); u.q[3] = *(const uint4*)(kp + 24);
      float mx = -1e30f;
      for (int d = 0; d < 32; ++d) mx = fmaxf(mx, bf2f(u.hh[d]));
      float ev[32], s = 0.f;
      for (int d = 0; d < 32; ++d) { ev[d] = __expf(bf2f(u.hh[d]) - mx); s += ev[d]; }
      float inv = 1.f / s;
      for (int d = 0; d < 32; ++d) {
        int laneA = (d & 15) + 16 * lgA;
        sKA[(((h * 4 + kstep) * 2 + (d >> 4)) << 9) + (laneA << 4) + iA] =
            f2bf1(ev[d] * inv);
      }
    }
    {  // v raw bits -> WMMA B-layout staging
      const unsigned short* vp = sT + j * QP + 256 + h * 32;
      u.q[0] = *(const uint4*)vp;        u.q[1] = *(const uint4*)(vp + 8);
      u.q[2] = *(const uint4*)(vp + 16); u.q[3] = *(const uint4*)(vp + 24);
      for (int e = 0; e < 32; ++e) {
        int laneB = (e & 15) + 16 * lgB;
        sVB[(((h * 4 + kstep) * 2 + (e >> 4)) << 9) + (laneB << 4) + iB] = u.hh[e];
      }
    }
  }
  __syncthreads();

  // ctx_p[h][d][e] = sum_j k'[d,j] v[e,j]  via WMMA, K = 128 positions
  {
    const int h = wave >> 1, m2 = wave & 1;
    float* cp = ctxPartial + (size_t)(b * NTILES + tile) * 4096 + h * 1024;
    for (int n2 = 0; n2 < 2; ++n2) {
      v8f acc = {0.f, 0.f, 0.f, 0.f, 0.f, 0.f, 0.f, 0.f};
      for (int ks = 0; ks < 4; ++ks) {
        BfPack a, bm;
        const unsigned short* ap = sKA + (((h * 4 + ks) * 2 + m2) << 9) + (lane << 4);
        const unsigned short* bp = sVB + (((h * 4 + ks) * 2 + n2) << 9) + (lane << 4);
        a.u.lo  = *(const uint4*)ap;  a.u.hi  = *(const uint4*)(ap + 8);
        bm.u.lo = *(const uint4*)bp;  bm.u.hi = *(const uint4*)(bp + 8);
        acc = __builtin_amdgcn_wmma_f32_16x16x32_bf16(false, a.v, false, bm.v,
                                                      (short)0, acc, false, false);
      }
#pragma unroll
      for (int r = 0; r < 8; ++r) {
        int d = m2 * 16 + r + 8 * lg;
        cp[d * 32 + n2 * 16 + lr] = acc[r];
      }
    }
  }
}

// ---------------------------------------------------------------------------
// K2: deterministic reduction of context partials over tiles
// ---------------------------------------------------------------------------
__global__ __launch_bounds__(256) void k_ctx(const float* __restrict__ cp,
                                             float* __restrict__ ctx) {
  int g = blockIdx.x * 256 + threadIdx.x;   // 16*4096 total
  int b = g >> 12, idx = g & 4095;
  const float* p = cp + (size_t)b * NTILES * 4096 + idx;
  float s = 0.f;
  for (int t = 0; t < NTILES; ++t) s += p[(size_t)t * 4096];
  ctx[g] = s;
}

// ---------------------------------------------------------------------------
// K3: async-copy q' tile to LDS; attn = ctx^T x q' (WMMA); y = Wout x attn + b
//     (WMMA) -> d_out; per-tile channel sum/sumsq partials
// ---------------------------------------------------------------------------
__global__ __launch_bounds__(256) void k_out(const unsigned short* __restrict__ Qp,
                                             const float* __restrict__ ctx,
                                             const float* __restrict__ wout,
                                             const float* __restrict__ bout,
                                             float* __restrict__ out,
                                             float* __restrict__ sumP,
                                             float* __restrict__ sqP) {
  extern __shared__ char smem[];
  unsigned short* sCT = (unsigned short*)smem;              // [h][e][d]  8KB
  unsigned short* sQ  = (unsigned short*)(smem + 8192);     // [n][hid]  32KB
  unsigned short* sAt = (unsigned short*)(smem + 40960);    // [n][e]    32KB
  unsigned short* sWo = (unsigned short*)(smem + 73728);    // 64*128    16KB
  float* sBo  = (float*)(smem + 90112);                     // 64
  float* sY   = (float*)(smem + 90368);                     // [n][ch] f32 32KB
  float* sRed = (float*)(smem + 123136);                    // 2*64*4

  const int tid = threadIdx.x;
  const int b = blockIdx.y, tile = blockIdx.x;
  const int n0 = tile * NT;

  // q' tile is raw bf16 in global: async DMA straight into LDS (no VGPR trip)
  {
    const unsigned lds0 = (unsigned)(uintptr_t)(void*)sQ;
    const unsigned short* gbase = Qp + (size_t)(b * NPIX + n0) * HIDD;
    for (int i = tid; i < (NT * HIDD) / 8; i += 256) {   // 2048 x 16B
      unsigned ldsa = lds0 + (unsigned)(i * 16);
      const unsigned short* g = gbase + i * 8;
      asm volatile("global_load_async_to_lds_b128 %0, %1, off"
                   :: "v"(ldsa), "v"(g) : "memory");
    }
  }

  for (int i = tid; i < (64 * HIDD) / 2; i += 256) {    // pairwise convert Wout
    float2 w2 = *(const float2*)(wout + 2 * i);
    ((unsigned*)sWo)[i] = cvt_pk_bf16(w2.x, w2.y);
  }
  if (tid < 64) sBo[tid] = bout[tid];
  for (int i = tid; i < 4096; i += 256) {
    int h = i >> 10, e = (i >> 5) & 31, d = i & 31;       // sCT = ctx transposed
    sCT[i] = f2bf1(ctx[b * 4096 + h * 1024 + d * 32 + e]);
  }
  asm volatile("s_wait_asynccnt 0" ::: "memory");
  __syncthreads();

  const int wave = tid >> 5, lane = tid & 31;
  const int lg = lane >> 4, lr = lane & 15;
  const int nb = wave * 16;

  for (int mt = 0; mt < 8; ++mt) {          // attn[e,n] per head, K = d = 32
    int h = mt >> 1;
    v8f acc = {0.f, 0.f, 0.f, 0.f, 0.f, 0.f, 0.f, 0.f};
    BfPack a, bm;
    const unsigned short* ar = sCT + h * 1024 + ((mt & 1) * 16 + lr) * 32;
    a.u.lo  = *(const uint4*)(ar + 8 * lg);
    a.u.hi  = *(const uint4*)(ar + 16 + 8 * lg);
    const unsigned short* br = sQ + (nb + lr) * HIDD + h * 32;
    bm.u.lo = *(const uint4*)(br + 8 * lg);
    bm.u.hi = *(const uint4*)(br + 16 + 8 * lg);
    acc = __builtin_amdgcn_wmma_f32_16x16x32_bf16(false, a.v, false, bm.v,
                                                  (short)0, acc, false, false);
    uint4 pk;
    pk.x = cvt_pk_bf16(acc[0], acc[1]); pk.y = cvt_pk_bf16(acc[2], acc[3]);
    pk.z = cvt_pk_bf16(acc[4], acc[5]); pk.w = cvt_pk_bf16(acc[6], acc[7]);
    *(uint4*)(sAt + (nb + lr) * HIDD + mt * 16 + 8 * lg) = pk;
  }
  __syncthreads();

  for (int mt = 0; mt < 4; ++mt) {          // y = Wout x attn, K = 128
    v8f acc = {0.f, 0.f, 0.f, 0.f, 0.f, 0.f, 0.f, 0.f};
    const unsigned short* ar = sWo + (mt * 16 + lr) * HIDD;
    const unsigned short* br = sAt + (nb + lr) * HIDD;
    for (int ks = 0; ks < 128; ks += 32) {
      BfPack a, bm;
      a.u.lo  = *(const uint4*)(ar + ks + 8 * lg);
      a.u.hi  = *(const uint4*)(ar + ks + 16 + 8 * lg);
      bm.u.lo = *(const uint4*)(br + ks + 8 * lg);
      bm.u.hi = *(const uint4*)(br + ks + 16 + 8 * lg);
      acc = __builtin_amdgcn_wmma_f32_16x16x32_bf16(false, a.v, false, bm.v,
                                                    (short)0, acc, false, false);
    }
    float y[8];
#pragma unroll
    for (int r = 0; r < 8; ++r) {
      int m = mt * 16 + r + 8 * lg;
      y[r] = acc[r] + sBo[m];
      out[(size_t)(b * CC + m) * NPIX + n0 + nb + lr] = y[r];
    }
    float4 f0 = {y[0], y[1], y[2], y[3]}, f1 = {y[4], y[5], y[6], y[7]};
    float* yp = sY + (nb + lr) * 64 + mt * 16 + 8 * lg;
    *(float4*)yp = f0; *(float4*)(yp + 4) = f1;
  }
  __syncthreads();

  {  // per-channel partial stats for BatchNorm (sY is [n][ch])
    int ch = tid & 63, q = tid >> 6;
    float s = 0.f, ss = 0.f;
    for (int j = q * 32; j < q * 32 + 32; ++j) {
      float v = sY[j * 64 + ch]; s += v; ss += v * v;
    }
    sRed[ch * 4 + q] = s;
    sRed[256 + ch * 4 + q] = ss;
  }
  __syncthreads();
  if (tid < 64) {
    float s  = sRed[tid * 4] + sRed[tid * 4 + 1] + sRed[tid * 4 + 2] + sRed[tid * 4 + 3];
    float ss = sRed[256 + tid * 4] + sRed[256 + tid * 4 + 1] +
               sRed[256 + tid * 4 + 2] + sRed[256 + tid * 4 + 3];
    sumP[(size_t)(b * NTILES + tile) * 64 + tid] = s;
    sqP[(size_t)(b * NTILES + tile) * 64 + tid]  = ss;
  }
}

// ---------------------------------------------------------------------------
// K4: final BN statistics
// ---------------------------------------------------------------------------
__global__ void k_stats(const float* __restrict__ sumP, const float* __restrict__ sqP,
                        float* __restrict__ stats) {
  int ch = threadIdx.x;                      // 64 threads
  float s = 0.f, ss = 0.f;
  for (int p = 0; p < BB * NTILES; ++p) { s += sumP[p * 64 + ch]; ss += sqP[p * 64 + ch]; }
  const float inv = 1.f / (float)(BB * NPIX);
  float mean = s * inv;
  float var  = ss * inv - mean * mean;
  stats[ch]      = mean;
  stats[64 + ch] = rsqrtf(var + 1e-5f);
}

// ---------------------------------------------------------------------------
// K5: in-place normalize + affine
// ---------------------------------------------------------------------------
__global__ __launch_bounds__(256) void k_bn(float* __restrict__ out,
                                            const float* __restrict__ stats,
                                            const float* __restrict__ gamma,
                                            const float* __restrict__ beta) {
  size_t idx = (size_t)blockIdx.x * 256 + threadIdx.x;
  int ch = (int)((idx >> 14) & 63);
  float v = out[idx];
  out[idx] = (v - stats[ch]) * stats[64 + ch] * gamma[ch] + beta[ch];
}

extern "C" void kernel_launch(void* const* d_in, const int* in_sizes, int n_in,
                              void* d_out, int out_size, void* d_ws, size_t ws_size,
                              hipStream_t stream) {
  const float* x     = (const float*)d_in[0];
  const float* wqkv  = (const float*)d_in[1];
  const float* wout  = (const float*)d_in[2];
  const float* bout  = (const float*)d_in[3];
  const float* gamma = (const float*)d_in[4];
  const float* beta  = (const float*)d_in[5];
  float* out = (float*)d_out;

  char* ws = (char*)d_ws;
  unsigned short* Qp = (unsigned short*)ws;                 // 64 MB bf16 q'
  float* ctxPartial  = (float*)(ws + 67108864);             // 32 MB
  float* ctx         = (float*)(ws + 100663296);            // 256 KB
  float* sumP        = (float*)(ws + 100925440);            // 512 KB
  float* sqP         = (float*)(ws + 101449728);            // 512 KB
  float* stats       = (float*)(ws + 101974016);            // 512 B

  dim3 grid1(NTILES, BB);
  k_qkv<<<grid1, 256, 231424, stream>>>(x, wqkv, Qp, ctxPartial);
  k_ctx<<<256, 256, 0, stream>>>(ctxPartial, ctx);
  k_out<<<grid1, 256, 125184, stream>>>(Qp, ctx, wout, bout, out, sumP, sqP);
  k_stats<<<1, 64, 0, stream>>>(sumP, sqP, stats);
  k_bn<<<65536, 256, 0, stream>>>(out, stats, gamma, beta);
}
